// RotaryMultiDotProductionAttention_88905823027990
// MI455X (gfx1250) — matrix operand order repaired and checked by reading, not compiled, not run on hardware
//
#include <hip/hip_runtime.h>

#define LSEQ 2048
#define DMODEL 1024
#define NHEAD 16
#define HDIM 64
#define BATCH 2
#define BN (BATCH * NHEAD)      // 32
#define MROWS (BATCH * LSEQ)    // 4096

typedef __attribute__((ext_vector_type(16))) __bf16 v16bf;
typedef __attribute__((ext_vector_type(8)))  float  v8f;
typedef __attribute__((ext_vector_type(4)))  unsigned int u32x4;

union Frag {
  v16bf v;
  u32x4 q[2];
};

// A operand (16x32 bf16): lane l<16 -> row l, K {koff..koff+15}, koff=0
// lane l>=16 -> row l-16, koff=16.  base must already include row*lda + k0 + koff.
__device__ __forceinline__ Frag load_fragA(const __bf16* base) {
  Frag f;
  const u32x4* p = (const u32x4*)base;
  f.q[0] = p[0];   // K koff .. koff+7
  f.q[1] = p[1];   // K koff+8 .. koff+15
  return f;
}

// B operand (32x16 bf16): lane l holds column l&15, 16 contiguous K values starting
// at k0 + 16*(l>>4).  base must already include col*ldb + k0 + 16*(l>>4).
__device__ __forceinline__ Frag load_fragB(const __bf16* base) {
  Frag f;
  const u32x4* p = (const u32x4*)base;
  f.q[0] = p[0];
  f.q[1] = p[1];
  return f;
}

__device__ __forceinline__ v8f wmma_bf16(Frag a, Frag b, v8f c) {
  return __builtin_amdgcn_wmma_f32_16x16x32_bf16(false, a.v, false, b.v,
                                                 (short)0, c, false, false);
}

// ---------------------------------------------------------------- convert
__global__ void cvt_f32_bf16(const float* __restrict__ in,
                             __bf16* __restrict__ out, int n) {
  int i = blockIdx.x * 256 + threadIdx.x;
  if (i < n) out[i] = (__bf16)in[i];
}

// ---------------------------------------------------------------- GEMM: C = X @ W^T + bias
// One wave computes a 16x64 output strip: A fragment reused by 4 independent WMMA
// accumulator chains (no WMMA->WMMA RAW hazard, better XDL occupancy).
// X: (MROWS x DMODEL) bf16 row-major, W: (DMODEL x DMODEL) bf16 row-major.
// mode 0: store bf16 head-major (b,n,l,h)    -> outB
// mode 2: store bf16 V-transposed (b,n,h,l)  -> outB
// mode 3: store fp32 (b,l,d)                 -> outF
__global__ void gemm16_proj(const __bf16* __restrict__ X,
                            const __bf16* __restrict__ W,
                            const float*  __restrict__ bias,
                            __bf16* __restrict__ outB,
                            float*  __restrict__ outF,
                            int mode) {
  const int lane  = threadIdx.x;           // 0..31 (wave32)
  const int r     = lane & 15;
  const int koffA = (lane >> 4) ? 16 : 0;
  const int koffB = (lane >> 4) * 16;
  const int n0 = blockIdx.x * 64;          // 64-wide strip
  const int m0 = blockIdx.y * 16;

  const __bf16* arow = X + (size_t)(m0 + r) * DMODEL + koffA;
  const __bf16* wrow = W + (size_t)(n0 + r) * DMODEL + koffB;
  const size_t wstep = (size_t)16 * DMODEL;   // 16 output columns

  v8f acc[4];
  #pragma unroll
  for (int j = 0; j < 4; ++j)
    #pragma unroll
    for (int v = 0; v < 8; ++v) acc[j][v] = 0.0f;

  #pragma unroll 2
  for (int k0 = 0; k0 < DMODEL; k0 += 32) {
    __builtin_prefetch(arow + k0 + 128, 0, 1);
    const Frag a = load_fragA(arow + k0);
    #pragma unroll
    for (int j = 0; j < 4; ++j) {
      const Frag b = load_fragB(wrow + (size_t)j * wstep + k0);
      acc[j] = wmma_bf16(a, b, acc[j]);
    }
  }

  const int rh = 8 * (lane >> 4);

  #pragma unroll
  for (int j = 0; j < 4; ++j) {
    const int col  = n0 + j * 16 + r;
    const float bv = bias[col];
    #pragma unroll
    for (int v = 0; v < 8; ++v) {
      const int m = m0 + v + rh;           // global row (b*L + l)
      const float val = acc[j][v] + bv;
      if (mode == 3) {
        outF[(size_t)m * DMODEL + col] = val;
      } else {
        const int b  = m >> 11;            // / LSEQ
        const int l  = m & (LSEQ - 1);
        const int nh = col >> 6;           // / HDIM
        const int h  = col & (HDIM - 1);
        if (mode == 2)
          outB[(((size_t)b * NHEAD + nh) * HDIM + h) * LSEQ + l] = (__bf16)val;
        else
          outB[(((size_t)b * NHEAD + nh) * LSEQ + l) * HDIM + h] = (__bf16)val;
      }
    }
  }
}

// ---------------------------------------------------------------- RoPE (in place, bf16 heads)
// buf layout (bn, l, h); pair i: e=x[2i], o=x[2i+1]; ang = l * 10000^{-i/32}
__global__ void rope_kernel(__bf16* __restrict__ buf) {
  const int idx = blockIdx.x * 256 + threadIdx.x;   // BN*LSEQ*32 threads
  const int i   = idx & 31;
  const int l   = (idx >> 5) & (LSEQ - 1);
  const int bn  = idx >> 16;
  if (bn >= BN) return;

  const float inv = __expf(-(float)i * 0.28782313662425576f);  // ln(10000)/32
  float s, c;
  __sincosf((float)l * inv, &s, &c);

  __bf16* p = buf + ((size_t)bn * LSEQ + l) * HDIM + 2 * i;
  const float e = (float)p[0];
  const float o = (float)p[1];
  p[0] = (__bf16)(e * c - o * s);
  p[1] = (__bf16)(o * c + e * s);
}

// ---------------------------------------------------------------- flash attention
// one wave per (b, n, 16-query-row block); keys streamed 32 at a time
__global__ void attn_kernel(const __bf16* __restrict__ Qh,   // (bn, l, h)
                            const __bf16* __restrict__ Kh,   // (bn, l, h)
                            const __bf16* __restrict__ Vt,   // (bn, h, l)
                            const int*    __restrict__ mask, // (b, l)
                            __bf16* __restrict__ Oa) {       // (b, l, d)
  __shared__ __bf16 Psh[16 * 32];

  const int lane  = threadIdx.x;
  const int r     = lane & 15;
  const int koffA = (lane >> 4) ? 16 : 0;
  const int koffB = (lane >> 4) * 16;
  const int rh    = 8 * (lane >> 4);

  const int tile = blockIdx.x;                 // 0 .. BN*(LSEQ/16)-1
  const int lb   = tile & (LSEQ / 16 - 1);
  const int bn   = tile >> 7;
  const int b    = bn >> 4;
  const int nh   = bn & 15;
  const int l0   = lb * 16;

  // Q fragments (A operand), rows l0..l0+15, K = h
  const __bf16* qbase = Qh + ((size_t)bn * LSEQ + l0 + r) * HDIM + koffA;
  const Frag qa0 = load_fragA(qbase);
  const Frag qa1 = load_fragA(qbase + 32);

  float mrun[8], srun[8];
  v8f acc[4];
  #pragma unroll
  for (int v = 0; v < 8; ++v) { mrun[v] = -3.0e38f; srun[v] = 0.0f; }
  #pragma unroll
  for (int t = 0; t < 4; ++t)
    #pragma unroll
    for (int v = 0; v < 8; ++v) acc[t][v] = 0.0f;

  const float scale = 0.03125f;                // 1/sqrt(DMODEL)

  for (int p0 = 0; p0 < LSEQ; p0 += 32) {
    // ---- scores: two 16x16 tiles, K over h (64 = 2 WMMA steps each)
    v8f S[2];
    #pragma unroll
    for (int t = 0; t < 2; ++t) {
      const __bf16* kb = Kh + ((size_t)bn * LSEQ + p0 + t * 16 + r) * HDIM + koffB;
      v8f z;
      #pragma unroll
      for (int v = 0; v < 8; ++v) z[v] = 0.0f;
      z = wmma_bf16(qa0, load_fragB(kb), z);
      S[t] = wmma_bf16(qa1, load_fragB(kb + 32), z);
    }

    // ---- scale + key mask (replace with -1e8)
    const int mv0 = mask[b * LSEQ + p0 + r];
    const int mv1 = mask[b * LSEQ + p0 + 16 + r];
    #pragma unroll
    for (int v = 0; v < 8; ++v) {
      S[0][v] = mv0 ? -1.0e8f : S[0][v] * scale;
      S[1][v] = mv1 ? -1.0e8f : S[1][v] * scale;
    }

    // ---- row max over 32 columns (cols live in lanes 0-15 / 16-31 groups)
    float mloc[8];
    #pragma unroll
    for (int v = 0; v < 8; ++v) mloc[v] = fmaxf(S[0][v], S[1][v]);
    #pragma unroll
    for (int d = 1; d < 16; d <<= 1)
      #pragma unroll
      for (int v = 0; v < 8; ++v)
        mloc[v] = fmaxf(mloc[v], __shfl_xor(mloc[v], d, 32));

    float alpha[8];
    #pragma unroll
    for (int v = 0; v < 8; ++v) {
      const float mn = fmaxf(mrun[v], mloc[v]);
      alpha[v] = __expf(mrun[v] - mn);
      mrun[v]  = mn;
    }

    // ---- probabilities + row sum
    float rsum[8];
    #pragma unroll
    for (int v = 0; v < 8; ++v) {
      S[0][v] = __expf(S[0][v] - mrun[v]);
      S[1][v] = __expf(S[1][v] - mrun[v]);
      rsum[v] = S[0][v] + S[1][v];
    }
    #pragma unroll
    for (int d = 1; d < 16; d <<= 1)
      #pragma unroll
      for (int v = 0; v < 8; ++v)
        rsum[v] += __shfl_xor(rsum[v], d, 32);
    #pragma unroll
    for (int v = 0; v < 8; ++v) srun[v] = srun[v] * alpha[v] + rsum[v];
    #pragma unroll
    for (int t = 0; t < 4; ++t)
      #pragma unroll
      for (int v = 0; v < 8; ++v) acc[t][v] *= alpha[v];

    // ---- relayout P (C-layout -> A-layout) through LDS (single wave: in-order)
    #pragma unroll
    for (int v = 0; v < 8; ++v) {
      Psh[(rh + v) * 32 + r]      = (__bf16)S[0][v];
      Psh[(rh + v) * 32 + 16 + r] = (__bf16)S[1][v];
    }
    const Frag pa = load_fragA(Psh + r * 32 + koffA);

    // ---- O += P @ V  (B columns = Vt rows, contiguous over p; 4 independent chains)
    #pragma unroll
    for (int th = 0; th < 4; ++th) {
      const __bf16* vb = Vt + ((size_t)bn * HDIM + th * 16 + r) * LSEQ + p0 + koffB;
      acc[th] = wmma_bf16(pa, load_fragB(vb), acc[th]);
    }
  }

  // ---- normalize + store bf16 into (b, l, d) for the output projection
  #pragma unroll
  for (int v = 0; v < 8; ++v) {
    const float inv = 1.0f / srun[v];
    const int l = l0 + v + rh;
    #pragma unroll
    for (int th = 0; th < 4; ++th) {
      const int col = nh * HDIM + th * 16 + r;
      Oa[((size_t)b * LSEQ + l) * DMODEL + col] = (__bf16)(acc[th][v] * inv);
    }
  }
}

// ----------------------------------------------------------------
extern "C" void kernel_launch(void* const* d_in, const int* in_sizes, int n_in,
                              void* d_out, int out_size, void* d_ws, size_t ws_size,
                              hipStream_t stream) {
  const float* q    = (const float*)d_in[0];
  const float* k    = (const float*)d_in[1];
  const float* v    = (const float*)d_in[2];
  const int*   mask = (const int*)  d_in[3];
  const float* Wq = (const float*)d_in[4];  const float* bq = (const float*)d_in[5];
  const float* Wk = (const float*)d_in[6];  const float* bk = (const float*)d_in[7];
  const float* Wv = (const float*)d_in[8];  const float* bv = (const float*)d_in[9];
  const float* Wo = (const float*)d_in[10]; const float* bo = (const float*)d_in[11];
  float* out = (float*)d_out;

  char* ws = (char*)d_ws;
  const size_t XE = (size_t)MROWS * DMODEL;       // 4M elems
  const size_t WE = (size_t)DMODEL * DMODEL;      // 1M elems
  __bf16* Xq  = (__bf16*)(ws);
  __bf16* Xk  = Xq + XE;
  __bf16* Xv  = Xk + XE;
  __bf16* Wqb = Xv + XE;
  __bf16* Wkb = Wqb + WE;
  __bf16* Wvb = Wkb + WE;
  __bf16* Wob = Wvb + WE;
  __bf16* Qh  = Wob + WE;
  __bf16* Kh  = Qh + XE;
  __bf16* Vt  = Kh + XE;
  __bf16* Oa  = Vt + XE;

  const int nX = (int)XE, nW = (int)WE;
  cvt_f32_bf16<<<(nX + 255) / 256, 256, 0, stream>>>(q, Xq, nX);
  cvt_f32_bf16<<<(nX + 255) / 256, 256, 0, stream>>>(k, Xk, nX);
  cvt_f32_bf16<<<(nX + 255) / 256, 256, 0, stream>>>(v, Xv, nX);
  cvt_f32_bf16<<<(nW + 255) / 256, 256, 0, stream>>>(Wq, Wqb, nW);
  cvt_f32_bf16<<<(nW + 255) / 256, 256, 0, stream>>>(Wk, Wkb, nW);
  cvt_f32_bf16<<<(nW + 255) / 256, 256, 0, stream>>>(Wv, Wvb, nW);
  cvt_f32_bf16<<<(nW + 255) / 256, 256, 0, stream>>>(Wo, Wob, nW);

  dim3 ggrid(DMODEL / 64, MROWS / 16);            // (16, 256)
  gemm16_proj<<<ggrid, 32, 0, stream>>>(Xq, Wqb, bq, Qh, nullptr, 0);
  gemm16_proj<<<ggrid, 32, 0, stream>>>(Xk, Wkb, bk, Kh, nullptr, 0);
  gemm16_proj<<<ggrid, 32, 0, stream>>>(Xv, Wvb, bv, Vt, nullptr, 2);

  const int ropeThreads = BN * LSEQ * 32;         // 2M
  rope_kernel<<<ropeThreads / 256, 256, 0, stream>>>(Qh);
  rope_kernel<<<ropeThreads / 256, 256, 0, stream>>>(Kh);

  attn_kernel<<<BN * (LSEQ / 16), 32, 0, stream>>>(Qh, Kh, Vt, mask, Oa);

  gemm16_proj<<<ggrid, 32, 0, stream>>>(Oa, Wob, bo, nullptr, out, 3);
}
